// Tagger_64733747085618
// MI455X (gfx1250) — matrix operand order repaired and checked by reading, not compiled, not run on hardware
//
#include <hip/hip_runtime.h>
#include <hip/hip_bf16.h>

// ---------------------------------------------------------------------------
// BiLSTM-CRF tagger for MI455X (gfx1250, wave32, WMMA + TDM).
//   B=64, T=1024, E=200 (pad 224), H=128/dir, 4H=512, HID=256, K=24 (pad 32)
//
//   1) k_prep   : weights -> f16 WMMA-B layout, fold gate biases
//   2) k_embed  : embedding gather -> X16 [T*B][224] f16
//   3) k_gx     : Gx = X @ W_ih^T + bias (f16 WMMA); OUTPUT IS STORED IN
//                 C-FRAGMENT ORDER, one contiguous 16KB block per
//                 (dir, t, btile) so the scan can DMA it with the TDM.
//   4) k_scan   : sequential LSTM scan, 8 persistent workgroups; W_hh
//                 fragments resident in VGPRs; next step's Gx block is
//                 prefetched into an LDS double buffer via tensor_load_to_lds
//                 (TENSORcnt), overlapped with the current step's WMMAs.
//   5) k_proj   : feats = H @ W_out^T + b_out (f16 WMMA)
//   6) k_viterbi: max-plus DP, one wave per sentence
// ---------------------------------------------------------------------------

#define BB   64
#define TT   1024
#define EE   200
#define EP   224
#define HH   128
#define G4   512
#define HID2 256
#define NT   24
#define NTP  32
#define VNEG -10000.0f

// Gx block: 8 scan-waves * 4 n-tiles * 32 lanes * 8 vals (f16) = 8192 elems
#define GXBLK 8192

typedef __attribute__((ext_vector_type(16))) _Float16 v16h;
typedef __attribute__((ext_vector_type(8)))  _Float16 v8h;
typedef __attribute__((ext_vector_type(8)))  float    v8f;

// --- WMMA fragment helpers (layouts per CDNA5 ISA 7.12.2) -------------------

__device__ __forceinline__ v16h make_frag(v8h lo, v8h hi) {
  return __builtin_shufflevector(lo, hi, 0,1,2,3,4,5,6,7,8,9,10,11,12,13,14,15);
}

// A fragment: 16x32 f16 tile from row-major [16][ld] source.
__device__ __forceinline__ v16h load_a_frag(const _Float16* a, int ld) {
  int lane = threadIdx.x & 31;
  int m = lane & 15, hi = lane >> 4;
  const _Float16* p = a + (size_t)m * ld + hi * 8;
  v8h x0 = *(const v8h*)(p);
  v8h x1 = *(const v8h*)(p + 16);
  return make_frag(x0, x1);
}

// B fragment: 32x16 f16 tile from row-major [32][ld] (K x N) source.
__device__ __forceinline__ v16h load_b_frag(const _Float16* b, int ld) {
  int lane = threadIdx.x & 31;
  int n = lane & 15, hi = lane >> 4;
  v16h r;
#pragma unroll
  for (int s = 0; s < 16; ++s) {
    int k = (s < 8) ? (8 * hi + s) : (16 + 8 * hi + (s - 8));
    r[s] = b[(size_t)k * ld + n];
  }
  return r;
}

__device__ __forceinline__ v8f wmma16(v16h a, v16h b, v8f c) {
  return __builtin_amdgcn_wmma_f32_16x16x32_f16(false, a, false, b, (short)0, c,
                                                false, false);
}

__device__ __forceinline__ float sigmoidf_(float x) {
  return 1.0f / (1.0f + __expf(-x));
}

// --- TDM 1-D copy (CDNA5 ISA ch.8 D# descriptor, VIMAGE TENSOR_LOAD_TO_LDS) -
// This toolchain exposes the 6-arg builtin:
//   (uint32x4 g0, int32x8 g1, int32x4 g2, int32x4 g3, int32x8 g4, i32 cpol)

#if __has_builtin(__builtin_amdgcn_tensor_load_to_lds)
#define USE_TDM 1
typedef unsigned int u32x4 __attribute__((ext_vector_type(4)));
typedef int          i32x8 __attribute__((ext_vector_type(8)));
typedef int          i32x4 __attribute__((ext_vector_type(4)));

// 1-D tile, data_size = 8 bytes, tile_dim0 = bytes/8 elements (<= 65535)
__device__ __forceinline__ void tdm_load_1d(const void* gaddr, void* laddr,
                                            unsigned int bytes) {
  unsigned long long ga = (unsigned long long)gaddr;
  unsigned int lds = (unsigned int)(unsigned long long)laddr;
  unsigned int el = bytes >> 3;
  u32x4 g0;
  g0[0] = 1u;                                           // count=1, user mode
  g0[1] = lds;                                          // lds_addr
  g0[2] = (unsigned int)(ga & 0xffffffffu);             // global_addr lo
  g0[3] = (unsigned int)((ga >> 32) & 0x1ffffffu) | (2u << 30);  // hi | type=2
  i32x8 g1;
  g1[0] = (3 << 16);                   // wg_mask=0, data_size=3 (8B), no flags
  g1[1] = (int)((el & 0xffffu) << 16); // tensor_dim0[15:0] @ bits 63:48
  g1[2] = (int)((el >> 16) & 0xffffu) | (1 << 16);  // dim0 hi | tensor_dim1=1
  g1[3] = (int)((el & 0xffffu) << 16); // tile_dim0 @ bits 127:112
  g1[4] = 0;                           // tile_dim1/2 unused
  g1[5] = (int)el;                     // tensor_dim0_stride lo
  g1[6] = 0;
  g1[7] = 0;
  i32x4 z4 = {0, 0, 0, 0};
  i32x8 z8 = {0, 0, 0, 0, 0, 0, 0, 0};
  __builtin_amdgcn_tensor_load_to_lds(g0, g1, z4, z4, z8, 0);
}
#else
#define USE_TDM 0
#endif

// --- 1) weight prep ---------------------------------------------------------

__global__ void k_prep(const float* WihF, const float* WhhF,
                       const float* bihF, const float* bhhF,
                       const float* WihB, const float* WhhB,
                       const float* bihB, const float* bhhB,
                       const float* Wout, const float* /*bout*/,
                       _Float16* WihT, _Float16* WhhT, _Float16* WoutT,
                       float* biasG) {
  int idx = blockIdx.x * blockDim.x + threadIdx.x;
  int stride = gridDim.x * blockDim.x;
  for (int i = idx; i < 2 * EP * G4; i += stride) {
    int d = i / (EP * G4), r = i % (EP * G4);
    int e = r / G4, g = r % G4;
    const float* W = d ? WihB : WihF;
    WihT[i] = (e < EE) ? (_Float16)W[(size_t)g * EE + e] : (_Float16)0.0f;
  }
  for (int i = idx; i < 2 * HH * G4; i += stride) {
    int d = i / (HH * G4), r = i % (HH * G4);
    int k = r / G4, g = r % G4;
    const float* W = d ? WhhB : WhhF;
    WhhT[i] = (_Float16)W[(size_t)g * HH + k];
  }
  for (int i = idx; i < HID2 * NTP; i += stride) {
    int k = i / NTP, n = i % NTP;
    WoutT[i] = (n < NT) ? (_Float16)Wout[(size_t)n * HID2 + k] : (_Float16)0.0f;
  }
  for (int i = idx; i < 2 * G4; i += stride) {
    int d = i / G4, g = i % G4;
    biasG[i] = d ? (bihB[g] + bhhB[g]) : (bihF[g] + bhhF[g]);
  }
}

// --- 2) embedding gather + f16 cast ----------------------------------------

__global__ void k_embed(const int* __restrict__ sent,
                        const float* __restrict__ emb,
                        _Float16* __restrict__ X) {
  int tb = blockIdx.x;             // row = t*B + b
  int t = tb / BB, b = tb % BB;
  int tok = sent[(size_t)b * TT + t];
  const float* src = emb + (size_t)tok * EE;
  _Float16* dst = X + (size_t)tb * EP;
  for (int e = threadIdx.x; e < EP; e += blockDim.x)
    dst[e] = (e < EE) ? (_Float16)src[e] : (_Float16)0.0f;
}

// --- 3) input-gate GEMM -----------------------------------------------------
// Block = 8 waves; wave w owns 4 M-tiles (rows bx*512 + w*16 + it*128),
// cols [by*64, +64).  B fragments are loaded once per k-chunk and reused by
// all 4 M-tiles (16 independent WMMAs per k-chunk).
// Output stored fragment-major: block (dir,t,btile) -> [sw][nt][lane][r] f16.

__global__ __launch_bounds__(256) void k_gx(const _Float16* __restrict__ X,
                                            const _Float16* __restrict__ WihT,
                                            const float* __restrict__ biasG,
                                            _Float16* __restrict__ Gx) {
  int dir = blockIdx.z;
  const _Float16* Wt = WihT + (size_t)dir * EP * G4;
  const float* bias = biasG + dir * G4;
  _Float16* gxd = Gx + (size_t)dir * TT * 4 * GXBLK;

  int wave = threadIdx.x >> 5, lane = threadIdx.x & 31;
  int sw = blockIdx.y;             // scan-wave / n-block index
  int n0 = sw * 64;
  int mbase = blockIdx.x * 512 + wave * 16;

  v8f acc[4][4];
#pragma unroll
  for (int it = 0; it < 4; ++it)
#pragma unroll
    for (int nt = 0; nt < 4; ++nt)
#pragma unroll
      for (int r = 0; r < 8; ++r) acc[it][nt][r] = 0.0f;

  for (int kc = 0; kc < EP / 32; ++kc) {
    v16h bfr[4];
#pragma unroll
    for (int nt = 0; nt < 4; ++nt)
      bfr[nt] = load_b_frag(Wt + (size_t)(kc * 32) * G4 + n0 + nt * 16, G4);
#pragma unroll
    for (int it = 0; it < 4; ++it) {
      v16h a = load_a_frag(X + (size_t)(mbase + it * 128) * EP + kc * 32, EP);
#pragma unroll
      for (int nt = 0; nt < 4; ++nt)
        acc[it][nt] = wmma16(a, bfr[nt], acc[it][nt]);
    }
  }

#pragma unroll
  for (int it = 0; it < 4; ++it) {
    int m0 = mbase + it * 128;           // row base = t*B + b
    int t = m0 >> 6, btile = (m0 >> 4) & 3;
    _Float16* dst = gxd + ((size_t)t * 4 + btile) * GXBLK;
#pragma unroll
    for (int nt = 0; nt < 4; ++nt) {
      float bn = bias[n0 + nt * 16 + (lane & 15)];
      v8h out;
#pragma unroll
      for (int r = 0; r < 8; ++r) out[r] = (_Float16)(acc[it][nt][r] + bn);
      *(v8h*)(dst + (((size_t)sw * 4 + nt) * 32 + lane) * 8) = out;
    }
  }
}

// --- 4) sequential LSTM scan ------------------------------------------------
// 8 blocks = 2 dir x 4 batch-tiles of 16 rows.  Wave w owns gate cols
// [64w, 64w+64) and keeps its 16 W_hh B-fragments in VGPRs.  The 16KB Gx
// block for step s+1 is DMA'd by the TDM into an LDS double buffer while
// step s computes; synchronization = s_wait_tensorcnt + per-step barrier.

__global__ __launch_bounds__(256) void k_scan(const _Float16* __restrict__ WhhT,
                                              const _Float16* __restrict__ Gx,
                                              const float* __restrict__ h0,
                                              const float* __restrict__ c0,
                                              _Float16* __restrict__ Hout) {
  int dir = blockIdx.x >> 2;
  int btile = blockIdx.x & 3;
  int b0 = btile * 16;
  int wave = threadIdx.x >> 5, lane = threadIdx.x & 31;
  const _Float16* Wt = WhhT + (size_t)dir * HH * G4;
  const _Float16* gxd = Gx + ((size_t)dir * TT + 0) * 4 * GXBLK;

  __shared__ float sGates[16 * G4];                     // 32 KB
  __shared__ float sC[16 * HH];                         //  8 KB
  __shared__ __align__(16) _Float16 sH[16 * HH];        //  4 KB
#if USE_TDM
  __shared__ __align__(16) _Float16 sGx[2][GXBLK];      // 32 KB double buffer
#endif

  for (int i = threadIdx.x; i < 16 * HH; i += blockDim.x) {
    int r = i / HH, hc = i % HH;
    size_t src = ((size_t)dir * BB + b0 + r) * HH + hc;
    sC[i] = c0[src];
    sH[i] = (_Float16)h0[src];
  }

  int n0 = wave * 64;
  v16h wb[4][4];
#pragma unroll
  for (int kc = 0; kc < 4; ++kc)
#pragma unroll
    for (int nt = 0; nt < 4; ++nt)
      wb[kc][nt] = load_b_frag(Wt + (size_t)(kc * 32) * G4 + n0 + nt * 16, G4);

#if USE_TDM
  if (wave == 0) {
    int tg0 = dir ? (TT - 1) : 0;
    tdm_load_1d(gxd + ((size_t)tg0 * 4 + btile) * GXBLK, &sGx[0][0],
                GXBLK * 2);
    __builtin_amdgcn_s_wait_tensorcnt(0);
  }
#endif
  __syncthreads();

  for (int s = 0; s < TT; ++s) {
    int tg = dir ? (TT - 1 - s) : s;

#if USE_TDM
    if (wave == 0 && s + 1 < TT) {
      int tgn = dir ? (TT - 2 - s) : (s + 1);
      tdm_load_1d(gxd + ((size_t)tgn * 4 + btile) * GXBLK,
                  &sGx[(s + 1) & 1][0], GXBLK * 2);
    }
    const _Float16* blk = &sGx[s & 1][0];
#else
    const _Float16* blk = gxd + ((size_t)tg * 4 + btile) * GXBLK;
#endif

    v8f acc[4];
#pragma unroll
    for (int nt = 0; nt < 4; ++nt) {
      v8h g = *(const v8h*)(blk + (((size_t)wave * 4 + nt) * 32 + lane) * 8);
#pragma unroll
      for (int r = 0; r < 8; ++r) acc[nt][r] = (float)g[r];
    }
#pragma unroll
    for (int kc = 0; kc < 4; ++kc) {
      v16h a = load_a_frag(sH + kc * 32, HH);
#pragma unroll
      for (int nt = 0; nt < 4; ++nt)
        acc[nt] = wmma16(a, wb[kc][nt], acc[nt]);
    }
#pragma unroll
    for (int nt = 0; nt < 4; ++nt) {
      int n = n0 + nt * 16 + (lane & 15);
#pragma unroll
      for (int r = 0; r < 8; ++r)
        sGates[(r + 8 * (lane >> 4)) * G4 + n] = acc[nt][r];
    }
    __syncthreads();   // h reads + gate writes complete

    for (int i = threadIdx.x; i < 16 * HH; i += blockDim.x) {
      int r = i / HH, hc = i % HH;
      float gi = sigmoidf_(sGates[r * G4 + hc]);
      float gf = sigmoidf_(sGates[r * G4 + HH + hc]);
      float gg = tanhf(sGates[r * G4 + 2 * HH + hc]);
      float go = sigmoidf_(sGates[r * G4 + 3 * HH + hc]);
      float c = gf * sC[i] + gi * gg;
      sC[i] = c;
      float h = go * tanhf(c);
      sH[i] = (_Float16)h;
      Hout[((size_t)tg * BB + b0 + r) * HID2 + dir * HH + hc] = (_Float16)h;
    }
#if USE_TDM
    if (wave == 0) __builtin_amdgcn_s_wait_tensorcnt(0);
#endif
    __syncthreads();   // h/c (+ next Gx buffer) published
  }
}

// --- 5) projection: feats = H @ WoutT + b_out ------------------------------

__global__ __launch_bounds__(256) void k_proj(const _Float16* __restrict__ Hbuf,
                                              const _Float16* __restrict__ WoutT,
                                              const float* __restrict__ bout,
                                              float* __restrict__ feats) {
  int wave = threadIdx.x >> 5, lane = threadIdx.x & 31;
  int m0 = blockIdx.x * 128 + wave * 16;

  v8f acc[2];
#pragma unroll
  for (int nt = 0; nt < 2; ++nt)
#pragma unroll
    for (int r = 0; r < 8; ++r) acc[nt][r] = 0.0f;

  for (int kc = 0; kc < HID2 / 32; ++kc) {
    v16h a = load_a_frag(Hbuf + (size_t)m0 * HID2 + kc * 32, HID2);
#pragma unroll
    for (int nt = 0; nt < 2; ++nt) {
      v16h bf = load_b_frag(WoutT + (size_t)(kc * 32) * NTP + nt * 16, NTP);
      acc[nt] = wmma16(a, bf, acc[nt]);
    }
  }
#pragma unroll
  for (int nt = 0; nt < 2; ++nt) {
    int n = nt * 16 + (lane & 15);
    float bn = (n < NT) ? bout[n] : 0.0f;
#pragma unroll
    for (int r = 0; r < 8; ++r) {
      int m = m0 + r + 8 * (lane >> 4);
      feats[(size_t)m * NTP + n] = acc[nt][r] + bn;
    }
  }
}

// --- 6) Viterbi: one wave per sentence -------------------------------------

__global__ void k_viterbi(const float* __restrict__ feats,
                          const float* __restrict__ trans,
                          unsigned char* __restrict__ bptr,
                          float* __restrict__ score_out,
                          float* __restrict__ tags_out) {
  int b = blockIdx.x;
  int lane = threadIdx.x;
  __shared__ float sT[NT * NT];
  for (int i = lane; i < NT * NT; i += 32) sT[i] = trans[i];
  __syncthreads();

  float v = (lane == 0) ? 0.0f : VNEG;     // START = 0
  for (int t = 0; t < TT; ++t) {
    float best = -3.4e38f;
    int bp = 0;
#pragma unroll
    for (int p = 0; p < NT; ++p) {
      float sv = __shfl(v, p, 32);
      float sc = sv + ((lane < NT) ? sT[lane * NT + p] : VNEG);
      if (sc > best) { best = sc; bp = p; }
    }
    float ft = (lane < NT) ? feats[((size_t)t * BB + b) * NTP + lane] : 0.0f;
    v = best + ft;
    if (lane < NT) bptr[((size_t)b * TT + t) * NT + lane] = (unsigned char)bp;
  }
  float tv = (lane < NT) ? v + sT[1 * NT + lane] : -3.4e38f;  // STOP = 1
  float best = -3.4e38f;
  int arg = 0;
#pragma unroll
  for (int p = 0; p < NT; ++p) {
    float sv = __shfl(tv, p, 32);
    if (sv > best) { best = sv; arg = p; }
  }
  if (lane == 0) {
    score_out[b] = best;
    int tag = arg;
    for (int t = TT - 1; t >= 0; --t) {
      tags_out[(size_t)b * TT + t] = (float)tag;
      tag = (int)bptr[((size_t)b * TT + t) * NT + tag];
    }
  }
}

// ---------------------------------------------------------------------------

extern "C" void kernel_launch(void* const* d_in, const int* in_sizes, int n_in,
                              void* d_out, int out_size, void* d_ws,
                              size_t ws_size, hipStream_t stream) {
  const int*   sent = (const int*)d_in[0];
  const float* emb  = (const float*)d_in[2];
  const float* WihF = (const float*)d_in[3];
  const float* WhhF = (const float*)d_in[4];
  const float* bihF = (const float*)d_in[5];
  const float* bhhF = (const float*)d_in[6];
  const float* WihB = (const float*)d_in[7];
  const float* WhhB = (const float*)d_in[8];
  const float* bihB = (const float*)d_in[9];
  const float* bhhB = (const float*)d_in[10];
  const float* h0   = (const float*)d_in[11];
  const float* c0   = (const float*)d_in[12];
  const float* Wout = (const float*)d_in[13];
  const float* bout = (const float*)d_in[14];
  const float* trans= (const float*)d_in[15];

  char* ws = (char*)d_ws;
  size_t off = 0;
  auto carve = [&](size_t bytes) {
    void* p = ws + off;
    off = (off + bytes + 255) & ~(size_t)255;
    return p;
  };
  _Float16* X16   = (_Float16*)carve((size_t)TT * BB * EP * 2);
  _Float16* WihT  = (_Float16*)carve((size_t)2 * EP * G4 * 2);
  _Float16* WhhT  = (_Float16*)carve((size_t)2 * HH * G4 * 2);
  _Float16* WoutT = (_Float16*)carve((size_t)HID2 * NTP * 2);
  float*    biasG = (float*)   carve((size_t)2 * G4 * 4);
  _Float16* Gx    = (_Float16*)carve((size_t)2 * TT * 4 * GXBLK * 2);
  _Float16* Hbuf  = (_Float16*)carve((size_t)TT * BB * HID2 * 2);
  float*    feats = (float*)   carve((size_t)TT * BB * NTP * 4);
  unsigned char* bptr = (unsigned char*)carve((size_t)BB * TT * NT);

  float* score_out = (float*)d_out;        // [B]
  float* tags_out  = score_out + BB;       // [B*T]

  k_prep<<<256, 256, 0, stream>>>(WihF, WhhF, bihF, bhhF,
                                  WihB, WhhB, bihB, bhhB,
                                  Wout, bout, WihT, WhhT, WoutT, biasG);
  k_embed<<<TT * BB, 256, 0, stream>>>(sent, emb, X16);
  k_gx<<<dim3((TT * BB) / 512, G4 / 64, 2), 256, 0, stream>>>(X16, WihT, biasG,
                                                              Gx);
  k_scan<<<8, 256, 0, stream>>>(WhhT, Gx, h0, c0, Hbuf);
  k_proj<<<(TT * BB) / 128, 256, 0, stream>>>(Hbuf, WoutT, bout, feats);
  k_viterbi<<<BB, 32, 0, stream>>>(feats, trans, bptr, score_out, tags_out);
}
